// DPMeans_27779848471138
// MI455X (gfx1250) — matrix operand order repaired and checked by reading, not compiled, not run on hardware
//
#include <hip/hip_runtime.h>
#include <hip/hip_bf16.h>

#define NPTS 65536
#define DIM  1024
#define KMAX 64
#define LAMBDA_ 1000.0f
#define INF_ 1e30f
#define MAX_ITER 50

// workspace layout (float offsets)
#define OFF_X2     16
#define OFF_M2     (OFF_X2 + NPTS)
#define OFF_DMIN   (OFF_M2 + KMAX)
#define OFF_NEWSUM (OFF_DMIN + NPTS)
#define OFF_SUMS   (OFF_NEWSUM + DIM)
#define OFF_COUNTS (OFF_SUMS + KMAX*DIM)
#define OFF_Z      (OFF_COUNTS + KMAX)
#define OFF_MUBF   (OFF_Z + NPTS)          // __bf16[KMAX*DIM] = 128 KB
// ctl ints: [0]=done [1]=K [2]=it [3]=nfar [4]=create [5]=Kc
// ctl floats: ws[8]=prev_obj ws[9]=obj_partial

typedef __attribute__((ext_vector_type(16))) __bf16 v16bf;
typedef __attribute__((ext_vector_type(8)))  float  v8f;

// ---------------- init / reset ----------------
__global__ void k_reset(float* ws) {
  int idx = blockIdx.x * 256 + threadIdx.x;
  if (idx == 0) { ((int*)ws)[3] = 0; ((int*)ws)[4] = 0; ((int*)ws)[5] = 0; ws[9] = 0.f; }
  if (idx < DIM)       ws[OFF_NEWSUM + idx] = 0.f;
  if (idx < KMAX*DIM)  ws[OFF_SUMS   + idx] = 0.f;
  if (idx < KMAX)      ws[OFF_COUNTS + idx] = 0.f;
}

// per-point squared norm (one wave per point, wave32 shuffle reduce)
__global__ void k_x2(const float* __restrict__ X, float* ws) {
  int wave = threadIdx.x >> 5, lane = threadIdx.x & 31;
  int p = blockIdx.x * 4 + wave;
  const float* xr = X + (size_t)p * DIM;
  float s = 0.f;
  for (int d = lane; d < DIM; d += 32) { float v = xr[d]; s += v * v; }
  #pragma unroll
  for (int m = 16; m >= 1; m >>= 1) s += __shfl_xor(s, m, 32);
  if (lane == 0) ws[OFF_X2 + p] = s;
}

// column sum of X over (all | far) points into NEWSUM, via LDS partials
__global__ void k_colsum(const float* __restrict__ X, float* ws, int farMode) {
  int* ci = (int*)ws;
  if (farMode && (ci[0] || !ci[4])) return;
  __shared__ float fsum[DIM];
  int tid = threadIdx.x;
  for (int d = tid; d < DIM; d += 256) fsum[d] = 0.f;
  __syncthreads();
  int base = blockIdx.x * 64;
  bool any = false;
  for (int p = 0; p < 64; ++p) {
    int i = base + p;
    if (farMode && !(ws[OFF_DMIN + i] > LAMBDA_)) continue;  // uniform
    any = true;
    const float* xr = X + (size_t)i * DIM;
    for (int d = tid; d < DIM; d += 256) fsum[d] += xr[d];
  }
  __syncthreads();
  if (farMode && !any) return;  // uniform
  for (int d = tid; d < DIM; d += 256) atomicAdd(&ws[OFF_NEWSUM + d], fsum[d]);
}

__global__ void k_init_mu(float* mu, float* ws) {
  int idx = blockIdx.x * 256 + threadIdx.x;
  if (idx == 0) { int* ci = (int*)ws; ci[0] = 0; ci[1] = 1; ci[2] = 0; ws[8] = 0.f; }
  if (idx < KMAX * DIM) {
    int k = idx / DIM, d = idx - k * DIM;
    mu[idx] = (k == 0) ? ws[OFF_NEWSUM + d] * (1.0f / NPTS) : 0.f;
  }
}

// ---------------- per-iteration kernels ----------------
// m2[k] = |mu_k|^2 and pre-convert mu -> bf16 copy for WMMA B operand
__global__ void k_m2(const float* __restrict__ mu, float* ws) {
  if (((int*)ws)[0]) return;
  __shared__ float red[256];
  int k = blockIdx.x, tid = threadIdx.x;
  const float* mr = mu + (size_t)k * DIM;
  __bf16* mubf = (__bf16*)(ws + OFF_MUBF);
  float s = 0.f;
  for (int d = tid; d < DIM; d += 256) {
    float v = mr[d];
    s += v * v;
    mubf[(size_t)k * DIM + d] = (__bf16)v;      // hw cvt, RNE
  }
  red[tid] = s; __syncthreads();
  for (int off = 128; off >= 1; off >>= 1) { if (tid < off) red[tid] += red[tid + off]; __syncthreads(); }
  if (tid == 0) ws[OFF_M2 + k] = red[0];
}

// WMMA distance GEMM + min/argmin: each wave = 16 points x 64 clusters
__global__ void __launch_bounds__(128)
k_assign(const float* __restrict__ X, const float* __restrict__ mu, float* ws) {
  int* ci = (int*)ws;
  if (ci[0]) return;                       // uniform: EXEC stays full for WMMA
  const int Kcur = ci[1];
  int wave = threadIdx.x >> 5, lane = threadIdx.x & 31;
  int half = lane >> 4, col = lane & 15;
  int pb = blockIdx.x * 64 + wave * 16;
  v8f acc[4] = {};
  const float* xr = X + (size_t)(pb + col) * DIM;           // A row for this lane
  const __bf16* mubf = (const __bf16*)(ws + OFF_MUBF);

  for (int kt = 0; kt < DIM; kt += 32) {
    __builtin_prefetch(xr + kt + 64, 0, 0);
    v16bf a;
    {
      const float* p0 = xr + kt + 16 * half;                // K = 16*half + j
      const float* p1 = xr + kt + 16 * half + 8;            // K = 16*half + 8 + j
      #pragma unroll
      for (int j = 0; j < 8; ++j) a[j]     = (__bf16)p0[j];
      #pragma unroll
      for (int j = 0; j < 8; ++j) a[j + 8] = (__bf16)p1[j];
    }
    #pragma unroll
    for (int t = 0; t < 4; ++t) {
      int c = 16 * t + col;                                  // B column = cluster
      v16bf b = *(const v16bf*)(mubf + (size_t)c * DIM + kt + 16 * half); // K = 16*half + j
      acc[t] = __builtin_amdgcn_wmma_f32_16x16x32_bf16(
          false, a, false, b, (short)0, acc[t], false, false);
    }
  }

  int nfar_local = 0;
  #pragma unroll
  for (int v = 0; v < 8; ++v) {
    int pt = pb + v + 8 * half;                              // C layout: M = v + 8*half
    float px2 = ws[OFF_X2 + pt];
    float bd = INF_; int bc = 0;
    #pragma unroll
    for (int t = 0; t < 4; ++t) {
      int c = 16 * t + col;
      float dist = (c < Kcur) ? (px2 - 2.0f * acc[t][v] + ws[OFF_M2 + c]) : INF_;
      if (dist < bd) { bd = dist; bc = c; }
    }
    #pragma unroll
    for (int m = 1; m <= 8; m <<= 1) {                       // reduce within 16-lane half
      float od = __shfl_xor(bd, m, 32);
      int   oc = __shfl_xor(bc, m, 32);
      if (od < bd || (od == bd && oc < bc)) { bd = od; bc = oc; }
    }
    if (col == 0) {
      ((int*)ws)[OFF_Z + pt] = bc;
      ws[OFF_DMIN + pt] = bd;
      if (bd > LAMBDA_) nfar_local++;
    }
  }
  if (col == 0 && nfar_local) atomicAdd(&ci[3], nfar_local);
}

__global__ void k_pre(float* ws) {
  int* ci = (int*)ws;
  if (ci[0]) return;
  ci[4] = (ci[3] > 0) && (ci[1] < KMAX);
  ci[5] = ci[1] < (KMAX - 1) ? ci[1] : (KMAX - 1);
}

__global__ void k_newcenter(float* mu, float* ws) {
  int* ci = (int*)ws;
  if (ci[0] || !ci[4]) return;
  float cnt = fmaxf((float)ci[3], 1.0f);
  int d = blockIdx.x * 256 + threadIdx.x;
  if (d < DIM) mu[(size_t)ci[5] * DIM + d] = ws[OFF_NEWSUM + d] / cnt;
}

// reassign far points to new cluster, dvals, counts histogram, objective partial
__global__ void k_fixup(const float* __restrict__ X, const float* __restrict__ mu, float* ws) {
  int* ci = (int*)ws;
  if (ci[0]) return;
  int create = ci[4], Kc = ci[5];
  __shared__ float red[256];
  __shared__ float objacc;
  __shared__ int hist[KMAX];
  int tid = threadIdx.x;
  if (tid < KMAX) hist[tid] = 0;
  if (tid == 0) objacc = 0.f;
  __syncthreads();
  int base = blockIdx.x * 64;
  const float* nc = mu + (size_t)Kc * DIM;
  for (int p = 0; p < 64; ++p) {
    int i = base + p;
    float dm = ws[OFF_DMIN + i];
    bool cf = create && (dm > LAMBDA_);              // uniform per p
    float dval = dm;
    if (cf) {
      const float* xr = X + (size_t)i * DIM;
      float s = 0.f;
      for (int d = tid; d < DIM; d += 256) { float df = xr[d] - nc[d]; s += df * df; }
      red[tid] = s; __syncthreads();
      for (int off = 128; off >= 1; off >>= 1) { if (tid < off) red[tid] += red[tid + off]; __syncthreads(); }
      dval = red[0];
      __syncthreads();
    }
    if (tid == 0) {
      int zi = ((int*)ws)[OFF_Z + i];
      if (cf) { zi = Kc; ((int*)ws)[OFF_Z + i] = Kc; }
      hist[zi]++;
      objacc += dval;
    }
  }
  __syncthreads();
  if (tid < KMAX && hist[tid]) atomicAdd(&ws[OFF_COUNTS + tid], (float)hist[tid]);
  if (tid == 0) atomicAdd(&ws[9], objacc);
}

// segment sums = one-hot(Z)^T @ X as a WMMA GEMM; wave = 64 clusters x 16 dims x 1024 pts
// z chunk staged into LDS with CDNA5 async global->LDS DMA (ASYNCcnt)
__global__ void __launch_bounds__(128)
k_segsum(const float* __restrict__ X, float* ws) {
  int* ci = (int*)ws;
  if (ci[0]) return;
  __shared__ int zs[1024];
  int slice = blockIdx.x & 15;       // dim slice (64 dims)
  int chunk = blockIdx.x >> 4;       // point chunk (1024 points)
  int tid = threadIdx.x;
  int kbase0 = chunk * 1024;

  // async copy z[kbase0 .. +1024) -> LDS (generic ptr low 32 bits == LDS address)
  {
    unsigned lds_base = (unsigned)(uintptr_t)&zs[0];
    const int* gz = ((const int*)ws) + OFF_Z + kbase0;
    for (int q = tid; q < 1024; q += 128) {
      unsigned ldsa = lds_base + (unsigned)q * 4u;
      unsigned long long ga = (unsigned long long)(uintptr_t)(gz + q);
      asm volatile("global_load_async_to_lds_b32 %0, %1, off"
                   :: "v"(ldsa), "v"(ga) : "memory");
    }
    asm volatile("s_wait_asynccnt 0" ::: "memory");
  }
  __syncthreads();

  int wave = tid >> 5, lane = tid & 31;
  int half = lane >> 4, col = lane & 15;
  int dimbase = slice * 64 + wave * 16;
  const __bf16 ONE = (__bf16)1.0f, ZERO = (__bf16)0.0f;
  v8f acc[4] = {};
  for (int kk = 0; kk < 1024; kk += 32) {
    v16bf b;                                        // B[k][n] = X[point][dim]
    #pragma unroll
    for (int j = 0; j < 16; ++j) {
      int kp = kbase0 + kk + 8 * half + j + (j >= 8 ? 8 : 0);
      b[j] = (__bf16)X[(size_t)kp * DIM + dimbase + col];
    }
    #pragma unroll
    for (int m = 0; m < 4; ++m) {
      int cl = 16 * m + col;                        // A row = cluster
      v16bf a;
      #pragma unroll
      for (int j = 0; j < 8; ++j) a[j]     = (zs[kk + 8 * half + j]      == cl) ? ONE : ZERO;
      #pragma unroll
      for (int j = 0; j < 8; ++j) a[j + 8] = (zs[kk + 16 + 8 * half + j] == cl) ? ONE : ZERO;
      acc[m] = __builtin_amdgcn_wmma_f32_16x16x32_bf16(
          false, a, false, b, (short)0, acc[m], false, false);
    }
  }
  #pragma unroll
  for (int m = 0; m < 4; ++m)
    #pragma unroll
    for (int v = 0; v < 8; ++v) {
      int cl = 16 * m + v + 8 * half;
      atomicAdd(&ws[OFF_SUMS + cl * DIM + dimbase + col], acc[m][v]);
    }
}

__global__ void k_cupdate(float* mu, float* ws) {
  int* ci = (int*)ws;
  if (ci[0]) return;
  int idx = blockIdx.x * 256 + threadIdx.x;
  if (idx >= KMAX * DIM) return;
  int k = idx / DIM;
  float c = ws[OFF_COUNTS + k];
  if (c > 0.f) mu[idx] = ws[OFF_SUMS + idx] / c;
}

__global__ void k_post(float* ws) {
  int* ci = (int*)ws;
  if (ci[0]) return;
  int Kn = ci[1] + (ci[4] ? 1 : 0);
  float obj = ws[9] + LAMBDA_ * (float)Kn;
  int conv = (ci[2] > 0) && (fabsf(obj - ws[8]) < 1e-3f * obj);
  ci[1] = Kn; ci[2] = ci[2] + 1; ws[8] = obj;
  if (conv) ci[0] = 1;
}

extern "C" void kernel_launch(void* const* d_in, const int* in_sizes, int n_in,
                              void* d_out, int out_size, void* d_ws, size_t ws_size,
                              hipStream_t stream) {
  (void)in_sizes; (void)n_in; (void)out_size; (void)ws_size;
  const float* X = (const float*)d_in[0];
  float* mu = (float*)d_out;            // output IS mu: (1, 64, 1024) fp32
  float* ws = (float*)d_ws;

  // init: mu0 = mean(X), x2 = rowwise |X|^2, control block
  k_reset  <<<256,        256, 0, stream>>>(ws);
  k_x2     <<<NPTS / 4,   128, 0, stream>>>(X, ws);
  k_colsum <<<NPTS / 64,  256, 0, stream>>>(X, ws, 0);
  k_init_mu<<<256,        256, 0, stream>>>(mu, ws);

  for (int it = 0; it < MAX_ITER; ++it) {
    k_reset    <<<256,       256, 0, stream>>>(ws);
    k_m2       <<<KMAX,      256, 0, stream>>>(mu, ws);
    k_assign   <<<NPTS / 64, 128, 0, stream>>>(X, mu, ws);
    k_pre      <<<1,         1,   0, stream>>>(ws);
    k_colsum   <<<NPTS / 64, 256, 0, stream>>>(X, ws, 1);
    k_newcenter<<<4,         256, 0, stream>>>(mu, ws);
    k_fixup    <<<NPTS / 64, 256, 0, stream>>>(X, mu, ws);
    k_segsum   <<<1024,      128, 0, stream>>>(X, ws);
    k_cupdate  <<<256,       256, 0, stream>>>(mu, ws);
    k_post     <<<1,         1,   0, stream>>>(ws);
  }
}
